// Parallel_Graphormer_47021301956898
// MI455X (gfx1250) — matrix operand order repaired and checked by reading, not compiled, not run on hardware
//
#include <hip/hip_runtime.h>

// ---------------------------------------------------------------------------
// Problem constants (B=8, T=12, N=1000, D=256, H=8, R=32, HD=32, E=1024)
// ---------------------------------------------------------------------------
#define MTOK   96000          // B*T*N tokens
#define DDIM   256
#define EDIM   1024
#define HHEADS 8
#define HDHEAD 32
#define RRANK  32
#define NSEQ   1000
#define BTH    768            // B*T*H blocks

typedef __attribute__((ext_vector_type(16))) __bf16 v16bf;
typedef __attribute__((ext_vector_type(8)))  __bf16 v8bf;
typedef __attribute__((ext_vector_type(8)))  float  v8f;

__device__ __forceinline__ float sigm(float v) { return 1.0f / (1.0f + __expf(-v)); }

__device__ __forceinline__ v16bf concat8(v8bf lo, v8bf hi) {
  return __builtin_shufflevector(lo, hi, 0,1,2,3,4,5,6,7,8,9,10,11,12,13,14,15);
}

// A fragment (16x32 bf16): lane half selects K chunks {h*8..h*8+7} and {16+h*8..}
__device__ __forceinline__ v16bf load_afrag_lds(const __bf16* base, int row, int half) {
  const v8bf* p0 = (const v8bf*)(base + row * 32 + half * 8);
  const v8bf* p1 = (const v8bf*)(base + row * 32 + 16 + half * 8);
  return concat8(*p0, *p1);
}
__device__ __forceinline__ v16bf load_afrag_g(const __bf16* A, size_t row, int ldk,
                                              int k0, int half) {
  const v8bf* p0 = (const v8bf*)(A + row * (size_t)ldk + k0 + half * 8);
  const v8bf* p1 = (const v8bf*)(A + row * (size_t)ldk + k0 + 16 + half * 8);
  return concat8(*p0, *p1);
}
// B fragment (32x16 bf16) from K-contiguous ("transposed") weights Wt[n*ldk + k]:
// lane half selects K chunks {h*8..h*8+7} and {16+h*8..} (matches A fragment).
__device__ __forceinline__ v16bf load_bfrag(const __bf16* Wt, int n, int ldk,
                                            int k0, int half) {
  const v8bf* p0 = (const v8bf*)(Wt + (size_t)n * ldk + k0 + half * 8);
  const v8bf* p1 = (const v8bf*)(Wt + (size_t)n * ldk + k0 + 16 + half * 8);
  return concat8(*p0, *p1);
}

// CDNA5 async copy: one b128 per thread, global -> LDS, tracked by ASYNCcnt.
__device__ __forceinline__ void stage_a_tile(const __bf16* A, size_t lda, size_t mBase,
                                             int kb, __bf16* As, int tid) {
  int row = tid >> 2;            // 0..63
  int kc  = (tid & 3) * 8;       // 0,8,16,24
  const __bf16* g = A + (mBase + row) * lda + kb + kc;
  unsigned loff = (unsigned)(unsigned long long)(void*)(As + row * 32 + kc);
  asm volatile("global_load_async_to_lds_b128 %0, %1, off" :: "v"(loff), "v"(g) : "memory");
}
// Double-buffer waits: async loads complete in order, so waiting ASYNCcnt<=1
// with one freshly issued stage outstanding guarantees the previous stage landed.
__device__ __forceinline__ void wait_async_le1() {
  asm volatile("s_wait_asynccnt 0x1" ::: "memory");
}
__device__ __forceinline__ void wait_async_le0() {
  asm volatile("s_wait_asynccnt 0x0" ::: "memory");
}

// ---------------------------------------------------------------------------
// K0: weight conversion + transpose to bf16 (K-contiguous for B fragments)
// ---------------------------------------------------------------------------
__global__ __launch_bounds__(256) void k_conv(
    const float* __restrict__ W1, const float* __restrict__ W2,
    const float* __restrict__ W3, const float* __restrict__ Wq,
    const float* __restrict__ Wv, const float* __restrict__ key,
    __bf16* W1t, __bf16* W2t, __bf16* W3t, __bf16* Wqt, __bf16* Wvt, __bf16* keyT) {
  int i = blockIdx.x * 256 + threadIdx.x;
  if (i < DDIM * EDIM) {
    int n = i >> 8, k = i & 255;           // W1t/W2t: [E][D]
    W1t[i] = (__bf16)W1[(size_t)k * EDIM + n];
    W2t[i] = (__bf16)W2[(size_t)k * EDIM + n];
    int n3 = i >> 10, k3 = i & 1023;       // W3t: [D][E]
    W3t[i] = (__bf16)W3[(size_t)k3 * DDIM + n3];
  }
  if (i < DDIM * DDIM) {
    int n = i >> 8, k = i & 255;           // Wqt/Wvt: [D][D]
    Wqt[i] = (__bf16)Wq[(size_t)k * DDIM + n];
    Wvt[i] = (__bf16)Wv[(size_t)k * DDIM + n];
  }
  if (i < RRANK * HHEADS * HDHEAD) {       // keyT: [h][r][d], d contiguous
    int h = i >> 10, rem = i & 1023, r = rem >> 5, d = rem & 31;
    keyT[i] = (__bf16)key[(size_t)r * (HHEADS * HDHEAD) + h * HDHEAD + d];
  }
}

// ---------------------------------------------------------------------------
// K1: fused double RMS-norm.  One wave per token (32 lanes x 8 elems).
// ---------------------------------------------------------------------------
__global__ __launch_bounds__(256) void k_norm(
    const float* __restrict__ x, const float* __restrict__ z,
    const float* __restrict__ n1w, const float* __restrict__ n2w,
    const float* __restrict__ gamma, const float* __restrict__ delta,
    __bf16* __restrict__ x1b, __bf16* __restrict__ z1b) {
  int w = threadIdx.x >> 5, lane = threadIdx.x & 31;
  size_t m = (size_t)blockIdx.x * 8 + w;
  size_t base = m * DDIM + lane * 8;
  float v[8], ss = 0.f;
#pragma unroll
  for (int i = 0; i < 8; ++i) { v[i] = x[base + i]; ss += v[i] * v[i]; }
#pragma unroll
  for (int o = 16; o > 0; o >>= 1) ss += __shfl_xor(ss, o, 32);
  float rs = rsqrtf(ss * (1.f / DDIM) + 1e-6f);
  float sg = sigm(gamma[0]), sd = sigm(delta[0]);
  float t[8], ss2 = 0.f;
#pragma unroll
  for (int i = 0; i < 8; ++i) {
    float x1 = n1w[lane * 8 + i] * v[i] * rs;
    x1b[base + i] = (__bf16)x1;
    t[i] = sg * x1 + sd * z[base + i];
    ss2 += t[i] * t[i];
  }
#pragma unroll
  for (int o = 16; o > 0; o >>= 1) ss2 += __shfl_xor(ss2, o, 32);
  float rs2 = rsqrtf(ss2 * (1.f / DDIM) + 1e-6f);
#pragma unroll
  for (int i = 0; i < 8; ++i) z1b[base + i] = (__bf16)(n2w[lane * 8 + i] * t[i] * rs2);
}

// ---------------------------------------------------------------------------
// K2: fused SwiGLU-in:  h = silu(x1@W1+b1) * (x1@W2+b2)     (M x E, bf16)
// Block tile 64(M) x 128(N); 8 waves as 2x4; wave tile 32x32, two acc sets.
// A-tiles double-buffered in LDS via the async-copy engine.
// ---------------------------------------------------------------------------
__global__ __launch_bounds__(256) void k_ffn_in(
    const __bf16* __restrict__ x1b, const __bf16* __restrict__ W1t,
    const __bf16* __restrict__ W2t, const float* __restrict__ b1,
    const float* __restrict__ b2, __bf16* __restrict__ hb) {
  __shared__ __bf16 As[2][64 * 32];
  int tid = threadIdx.x;
  int w = tid >> 5, lane = tid & 31;
  int wr = w >> 2, wc = w & 3;
  int half = lane >> 4, l16 = lane & 15;
  size_t mBlk = (size_t)blockIdx.y * 64;
  int nBlk = blockIdx.x * 128;

  v8f zf = {0.f, 0.f, 0.f, 0.f, 0.f, 0.f, 0.f, 0.f};
  v8f acc1[2][2], acc2[2][2];
#pragma unroll
  for (int i = 0; i < 2; ++i)
#pragma unroll
    for (int j = 0; j < 2; ++j) { acc1[i][j] = zf; acc2[i][j] = zf; }

  stage_a_tile(x1b, DDIM, mBlk, 0, As[0], tid);
  int p = 0;
  for (int kb = 0; kb < DDIM; kb += 32, p ^= 1) {
    if (kb + 32 < DDIM) {
      stage_a_tile(x1b, DDIM, mBlk, kb + 32, As[p ^ 1], tid);
      __builtin_prefetch(W1t + (size_t)(nBlk + (tid >> 1)) * DDIM + kb + 32, 0, 1);
      wait_async_le1();
    } else {
      wait_async_le0();
    }
    __syncthreads();
#pragma unroll
    for (int fm = 0; fm < 2; ++fm) {
      v16bf a = load_afrag_lds(As[p], wr * 32 + fm * 16 + l16, half);
#pragma unroll
      for (int fn = 0; fn < 2; ++fn) {
        int n = nBlk + wc * 32 + fn * 16 + l16;
        v16bf bB1 = load_bfrag(W1t, n, DDIM, kb, half);
        v16bf bB2 = load_bfrag(W2t, n, DDIM, kb, half);
        acc1[fm][fn] = __builtin_amdgcn_wmma_f32_16x16x32_bf16(
            false, a, false, bB1, (short)0, acc1[fm][fn], false, false);
        acc2[fm][fn] = __builtin_amdgcn_wmma_f32_16x16x32_bf16(
            false, a, false, bB2, (short)0, acc2[fm][fn], false, false);
      }
    }
    __syncthreads();
  }
#pragma unroll
  for (int fm = 0; fm < 2; ++fm)
#pragma unroll
    for (int fn = 0; fn < 2; ++fn) {
      int n = nBlk + wc * 32 + fn * 16 + l16;
      float bb1 = b1[n], bb2 = b2[n];
#pragma unroll
      for (int i = 0; i < 8; ++i) {
        size_t mrow = mBlk + wr * 32 + fm * 16 + half * 8 + i;
        float u = acc1[fm][fn][i] + bb1;
        float g = acc2[fm][fn][i] + bb2;
        hb[mrow * EDIM + n] = (__bf16)((u * sigm(u)) * g);
      }
    }
}

// ---------------------------------------------------------------------------
// K3: ffn_out: d_out = sig(alpha) * (h @ W3 + b3)           (M x D, f32)
// ---------------------------------------------------------------------------
__global__ __launch_bounds__(256) void k_ffn_out(
    const __bf16* __restrict__ hb, const __bf16* __restrict__ W3t,
    const float* __restrict__ b3, const float* __restrict__ alpha,
    float* __restrict__ out) {
  __shared__ __bf16 As[2][64 * 32];
  int tid = threadIdx.x;
  int w = tid >> 5, lane = tid & 31;
  int wr = w >> 2, wc = w & 3;
  int half = lane >> 4, l16 = lane & 15;
  size_t mBlk = (size_t)blockIdx.y * 64;
  int nBlk = blockIdx.x * 128;

  v8f zf = {0.f, 0.f, 0.f, 0.f, 0.f, 0.f, 0.f, 0.f};
  v8f acc[2][2];
#pragma unroll
  for (int i = 0; i < 2; ++i)
#pragma unroll
    for (int j = 0; j < 2; ++j) acc[i][j] = zf;

  stage_a_tile(hb, EDIM, mBlk, 0, As[0], tid);
  int p = 0;
  for (int kb = 0; kb < EDIM; kb += 32, p ^= 1) {
    if (kb + 32 < EDIM) {
      stage_a_tile(hb, EDIM, mBlk, kb + 32, As[p ^ 1], tid);
      __builtin_prefetch(W3t + (size_t)(nBlk + (tid >> 1)) * EDIM + kb + 32, 0, 1);
      wait_async_le1();
    } else {
      wait_async_le0();
    }
    __syncthreads();
#pragma unroll
    for (int fm = 0; fm < 2; ++fm) {
      v16bf a = load_afrag_lds(As[p], wr * 32 + fm * 16 + l16, half);
#pragma unroll
      for (int fn = 0; fn < 2; ++fn) {
        int n = nBlk + wc * 32 + fn * 16 + l16;
        v16bf bB = load_bfrag(W3t, n, EDIM, kb, half);
        acc[fm][fn] = __builtin_amdgcn_wmma_f32_16x16x32_bf16(
            false, a, false, bB, (short)0, acc[fm][fn], false, false);
      }
    }
    __syncthreads();
  }
  float sa = sigm(alpha[0]);
#pragma unroll
  for (int fm = 0; fm < 2; ++fm)
#pragma unroll
    for (int fn = 0; fn < 2; ++fn) {
      int n = nBlk + wc * 32 + fn * 16 + l16;
      float bb = b3[n];
#pragma unroll
      for (int i = 0; i < 8; ++i) {
        size_t mrow = mBlk + wr * 32 + fm * 16 + half * 8 + i;
        out[mrow * DDIM + n] = sa * (acc[fm][fn][i] + bb);
      }
    }
}

// ---------------------------------------------------------------------------
// K4: generic (M x 256) @ (256 x 256) + bias -> bf16   (used for q and xv)
// ---------------------------------------------------------------------------
__global__ __launch_bounds__(256) void k_gemm_d(
    const __bf16* __restrict__ A, const __bf16* __restrict__ Wt,
    const float* __restrict__ bias, __bf16* __restrict__ outb) {
  __shared__ __bf16 As[2][64 * 32];
  int tid = threadIdx.x;
  int w = tid >> 5, lane = tid & 31;
  int wr = w >> 2, wc = w & 3;
  int half = lane >> 4, l16 = lane & 15;
  size_t mBlk = (size_t)blockIdx.y * 64;
  int nBlk = blockIdx.x * 128;

  v8f zf = {0.f, 0.f, 0.f, 0.f, 0.f, 0.f, 0.f, 0.f};
  v8f acc[2][2];
#pragma unroll
  for (int i = 0; i < 2; ++i)
#pragma unroll
    for (int j = 0; j < 2; ++j) acc[i][j] = zf;

  stage_a_tile(A, DDIM, mBlk, 0, As[0], tid);
  int p = 0;
  for (int kb = 0; kb < DDIM; kb += 32, p ^= 1) {
    if (kb + 32 < DDIM) {
      stage_a_tile(A, DDIM, mBlk, kb + 32, As[p ^ 1], tid);
      wait_async_le1();
    } else {
      wait_async_le0();
    }
    __syncthreads();
#pragma unroll
    for (int fm = 0; fm < 2; ++fm) {
      v16bf a = load_afrag_lds(As[p], wr * 32 + fm * 16 + l16, half);
#pragma unroll
      for (int fn = 0; fn < 2; ++fn) {
        int n = nBlk + wc * 32 + fn * 16 + l16;
        v16bf bB = load_bfrag(Wt, n, DDIM, kb, half);
        acc[fm][fn] = __builtin_amdgcn_wmma_f32_16x16x32_bf16(
            false, a, false, bB, (short)0, acc[fm][fn], false, false);
      }
    }
    __syncthreads();
  }
#pragma unroll
  for (int fm = 0; fm < 2; ++fm)
#pragma unroll
    for (int fn = 0; fn < 2; ++fn) {
      int n = nBlk + wc * 32 + fn * 16 + l16;
      float bb = bias[n];
#pragma unroll
      for (int i = 0; i < 8; ++i) {
        size_t mrow = mBlk + wr * 32 + fm * 16 + half * 8 + i;
        outb[mrow * DDIM + n] = (__bf16)(acc[fm][fn][i] + bb);
      }
    }
}

// ---------------------------------------------------------------------------
// K5: attn[m,h,r] = (q[m,h,:] . key[r,h,:]) / sqrt(HD)   via WMMA (K=32)
// Block: 8 waves = 8 heads, 32 tokens per block.
// ---------------------------------------------------------------------------
__global__ __launch_bounds__(256) void k_attn(
    const __bf16* __restrict__ qb, const __bf16* __restrict__ keyT,
    float* __restrict__ attn) {
  int tid = threadIdx.x;
  int h = tid >> 5, lane = tid & 31;
  int half = lane >> 4, l16 = lane & 15;
  size_t m0 = (size_t)blockIdx.x * 32;

  v8f zf = {0.f, 0.f, 0.f, 0.f, 0.f, 0.f, 0.f, 0.f};
  v8f acc[2][2];
#pragma unroll
  for (int i = 0; i < 2; ++i)
#pragma unroll
    for (int j = 0; j < 2; ++j) acc[i][j] = zf;

  const __bf16* kh = keyT + (size_t)h * (RRANK * HDHEAD);
#pragma unroll
  for (int fm = 0; fm < 2; ++fm) {
    v16bf a = load_afrag_g(qb, m0 + fm * 16 + l16, DDIM, h * HDHEAD, half);
#pragma unroll
    for (int fn = 0; fn < 2; ++fn) {
      v16bf bB = load_bfrag(kh, fn * 16 + l16, HDHEAD, 0, half);
      acc[fm][fn] = __builtin_amdgcn_wmma_f32_16x16x32_bf16(
          false, a, false, bB, (short)0, acc[fm][fn], false, false);
    }
  }
  const float scale = 0.17677669529663689f;  // 1/sqrt(32)
#pragma unroll
  for (int fm = 0; fm < 2; ++fm)
#pragma unroll
    for (int fn = 0; fn < 2; ++fn)
#pragma unroll
      for (int i = 0; i < 8; ++i) {
        size_t m = m0 + fm * 16 + half * 8 + i;
        attn[m * (HHEADS * RRANK) + h * RRANK + fn * 16 + l16] = acc[fm][fn][i] * scale;
      }
}

// ---------------------------------------------------------------------------
// K6: per-(b,t,h,r) online max/sum over n (softmax axis=2 stats)
// ---------------------------------------------------------------------------
__global__ __launch_bounds__(256) void k_colstats(
    const float* __restrict__ attn, float* __restrict__ cmax, float* __restrict__ csum) {
  int bth = blockIdx.x;
  int bt = bth >> 3, h = bth & 7;
  int r = threadIdx.x & 31, chunk = threadIdx.x >> 5;
  float m = -3.4e38f, s = 0.f;
  for (int n = chunk * 125; n < chunk * 125 + 125; ++n) {
    float a = attn[((size_t)bt * NSEQ + n) * (HHEADS * RRANK) + h * RRANK + r];
    float nm = fmaxf(m, a);
    s = s * __expf(m - nm) + __expf(a - nm);
    m = nm;
  }
  __shared__ float sm[8][32], ss[8][32];
  sm[chunk][r] = m;
  ss[chunk][r] = s;
  __syncthreads();
  if (threadIdx.x < 32) {
    int rr = threadIdx.x;
    float M = sm[0][rr], S = ss[0][rr];
#pragma unroll
    for (int c = 1; c < 8; ++c) {
      float m2 = sm[c][rr], s2 = ss[c][rr];
      float nm = fmaxf(M, m2);
      S = S * __expf(M - nm) + s2 * __expf(m2 - nm);
      M = nm;
    }
    cmax[(size_t)bth * 32 + rr] = M;
    csum[(size_t)bth * 32 + rr] = S;
  }
}

// ---------------------------------------------------------------------------
// K7: V1[r,d] = sum_n rowsoftmax(attn)[n,r] * xv[n,d]   per (b,t,h)
// ---------------------------------------------------------------------------
__global__ __launch_bounds__(256) void k_v1(
    const float* __restrict__ attn, const __bf16* __restrict__ xvb,
    float* __restrict__ V1) {
  int bth = blockIdx.x;
  int bt = bth >> 3, h = bth & 7;
  int w = threadIdx.x >> 5, lane = threadIdx.x & 31;
  __shared__ float acc_s[RRANK * HDHEAD];
  for (int i = threadIdx.x; i < RRANK * HDHEAD; i += 256) acc_s[i] = 0.f;
  __syncthreads();

  float acc[32];
#pragma unroll
  for (int d = 0; d < 32; ++d) acc[d] = 0.f;

  for (int n = w; n < NSEQ; n += 8) {
    size_t base = ((size_t)bt * NSEQ + n) * DDIM + h * HDHEAD;  // attn & xv share layout
    float a = attn[base + lane];
    float mx = a;
#pragma unroll
    for (int o = 16; o > 0; o >>= 1) mx = fmaxf(mx, __shfl_xor(mx, o, 32));
    float e = __expf(a - mx);
    float sum = e;
#pragma unroll
    for (int o = 16; o > 0; o >>= 1) sum += __shfl_xor(sum, o, 32);
    float p = e / sum;                    // P[n, r=lane]
    float xvv = (float)xvb[base + lane];  // xv[n, d=lane]
#pragma unroll
    for (int d = 0; d < 32; ++d) acc[d] += p * __shfl(xvv, d, 32);
  }
#pragma unroll
  for (int d = 0; d < 32; ++d) {
    int j = (d + w * 4) & 31;  // stagger to reduce LDS atomic conflicts
    atomicAdd(&acc_s[lane * 32 + j], acc[j]);
  }
  __syncthreads();
  for (int i = threadIdx.x; i < RRANK * HDHEAD; i += 256)
    V1[(size_t)bth * (RRANK * HDHEAD) + i] = acc_s[i];
}

// ---------------------------------------------------------------------------
// K8: out += sig(beta) * (sig(a_h)*xv + sig(b_h)*colsoftmax(attn)@V1)
// ---------------------------------------------------------------------------
__global__ __launch_bounds__(256) void k_combine(
    const float* __restrict__ attn, const __bf16* __restrict__ xvb,
    const float* __restrict__ V1, const float* __restrict__ cmax,
    const float* __restrict__ csum, const float* __restrict__ mha_a,
    const float* __restrict__ mha_b, const float* __restrict__ beta,
    float* __restrict__ out) {
  int bth = blockIdx.x;
  int bt = bth >> 3, h = bth & 7;
  __shared__ float v1s[RRANK * HDHEAD];
  __shared__ float cms[32], css[32];
  for (int i = threadIdx.x; i < RRANK * HDHEAD; i += 256)
    v1s[i] = V1[(size_t)bth * (RRANK * HDHEAD) + i];
  if (threadIdx.x < 32) {
    cms[threadIdx.x] = cmax[(size_t)bth * 32 + threadIdx.x];
    css[threadIdx.x] = csum[(size_t)bth * 32 + threadIdx.x];
  }
  __syncthreads();
  float sb = sigm(beta[0]);
  float sah = sigm(mha_a[h]);
  float sbh = sigm(mha_b[h]);
  int w = threadIdx.x >> 5, lane = threadIdx.x & 31;
  for (int n = w; n < NSEQ; n += 8) {
    size_t base = ((size_t)bt * NSEQ + n) * DDIM + h * HDHEAD;
    float a = attn[base + lane];
    float e = __expf(a - cms[lane]) / css[lane];  // Q[n, r=lane]
    float acc = 0.f;
#pragma unroll
    for (int r = 0; r < 32; ++r) acc += __shfl(e, r, 32) * v1s[r * 32 + lane];
    float xvv = (float)xvb[base + lane];
    out[base + lane] += sb * (sah * xvv + sbh * acc);
  }
}

// ---------------------------------------------------------------------------
// Host launcher
// ---------------------------------------------------------------------------
extern "C" void kernel_launch(void* const* d_in, const int* in_sizes, int n_in,
                              void* d_out, int out_size, void* d_ws, size_t ws_size,
                              hipStream_t stream) {
  const float* x     = (const float*)d_in[0];
  const float* z     = (const float*)d_in[1];
  const float* n1w   = (const float*)d_in[2];
  const float* n2w   = (const float*)d_in[3];
  const float* alpha = (const float*)d_in[4];
  const float* beta  = (const float*)d_in[5];
  const float* gamma = (const float*)d_in[6];
  const float* delta = (const float*)d_in[7];
  const float* Wq    = (const float*)d_in[8];
  const float* bq    = (const float*)d_in[9];
  const float* key   = (const float*)d_in[10];
  const float* Wv    = (const float*)d_in[11];
  const float* bv    = (const float*)d_in[12];
  const float* mha_a = (const float*)d_in[13];
  const float* mha_b = (const float*)d_in[14];
  const float* W1    = (const float*)d_in[15];
  const float* b1    = (const float*)d_in[16];
  const float* W2    = (const float*)d_in[17];
  const float* b2    = (const float*)d_in[18];
  const float* W3    = (const float*)d_in[19];
  const float* b3    = (const float*)d_in[20];
  float* out = (float*)d_out;
  char* ws = (char*)d_ws;

  size_t off = 0;
  auto alloc = [&](size_t bytes) {
    size_t o = off;
    off += (bytes + 255) & ~(size_t)255;
    return o;
  };
  const size_t M = MTOK;
  __bf16* x1b  = (__bf16*)(ws + alloc(M * DDIM * 2));
  __bf16* z1b  = (__bf16*)(ws + alloc(M * DDIM * 2));
  __bf16* qb   = (__bf16*)(ws + alloc(M * DDIM * 2));
  __bf16* xvb  = (__bf16*)(ws + alloc(M * DDIM * 2));
  __bf16* hb   = (__bf16*)(ws + alloc(M * EDIM * 2));
  float*  attn = (float*)(ws + alloc(M * HHEADS * RRANK * 4));
  float*  V1   = (float*)(ws + alloc((size_t)BTH * RRANK * HDHEAD * 4));
  float*  cmax = (float*)(ws + alloc((size_t)BTH * RRANK * 4));
  float*  csum = (float*)(ws + alloc((size_t)BTH * RRANK * 4));
  __bf16* W1t  = (__bf16*)(ws + alloc((size_t)DDIM * EDIM * 2));
  __bf16* W2t  = (__bf16*)(ws + alloc((size_t)DDIM * EDIM * 2));
  __bf16* W3t  = (__bf16*)(ws + alloc((size_t)EDIM * DDIM * 2));
  __bf16* Wqt  = (__bf16*)(ws + alloc((size_t)DDIM * DDIM * 2));
  __bf16* Wvt  = (__bf16*)(ws + alloc((size_t)DDIM * DDIM * 2));
  __bf16* keyT = (__bf16*)(ws + alloc((size_t)RRANK * HHEADS * HDHEAD * 2));

  k_conv<<<1024, 256, 0, stream>>>(W1, W2, W3, Wq, Wv, key, W1t, W2t, W3t, Wqt, Wvt, keyT);
  k_norm<<<MTOK / 8, 256, 0, stream>>>(x, z, n1w, n2w, gamma, delta, x1b, z1b);
  k_ffn_in<<<dim3(EDIM / 128, MTOK / 64), 256, 0, stream>>>(x1b, W1t, W2t, b1, b2, hb);
  k_ffn_out<<<dim3(DDIM / 128, MTOK / 64), 256, 0, stream>>>(hb, W3t, b3, alpha, out);
  k_gemm_d<<<dim3(DDIM / 128, MTOK / 64), 256, 0, stream>>>(z1b, Wqt, bq, qb);
  k_gemm_d<<<dim3(DDIM / 128, MTOK / 64), 256, 0, stream>>>(x1b, Wvt, bv, xvb);
  k_attn<<<MTOK / 32, 256, 0, stream>>>(qb, keyT, attn);
  k_colstats<<<BTH, 256, 0, stream>>>(attn, cmax, csum);
  k_v1<<<BTH, 256, 0, stream>>>(attn, xvb, V1);
  k_combine<<<BTH, 256, 0, stream>>>(attn, xvb, V1, cmax, csum, mha_a, mha_b, beta, out);
}
